// Transformer_75900662055217
// MI455X (gfx1250) — matrix operand order repaired and checked by reading, not compiled, not run on hardware
//
#include <hip/hip_runtime.h>
#include <math.h>

#define DEVI __device__ __forceinline__

typedef __attribute__((ext_vector_type(16))) __bf16 v16bf;
typedef __attribute__((ext_vector_type(8)))  float  v8f;

union Frag { v16bf v; uint4 u[2]; };

DEVI v8f zero8() { v8f z = {0.f,0.f,0.f,0.f,0.f,0.f,0.f,0.f}; return z; }

DEVI v8f wmma_bf16(v16bf a, v16bf b, v8f c) {
  // D = A(16x32 bf16) * B(32x16 bf16) + C(16x16 f32)
  return __builtin_amdgcn_wmma_f32_16x16x32_bf16(false, a, false, b, (short)0, c, false, false);
}

// Async global -> LDS copy of 16 bytes (CDNA5 GLOBAL_LOAD_ASYNC_TO_LDS_B128),
// tracked by ASYNCcnt.  lds_addr is the wave-relative LDS byte address
// (low 32 bits of the generic pointer; aperture bits live in [63:32]).
DEVI void async_b128(unsigned int lds_addr, const void* gaddr) {
  asm volatile("global_load_async_to_lds_b128 %0, %1, off"
               :: "v"(lds_addr), "v"((unsigned long long)(uintptr_t)gaddr)
               : "memory");
}
DEVI void wait_async0() { asm volatile("s_wait_asynccnt 0x0" ::: "memory"); }

// A fragment, 16x32 bf16, from row-major [M x K] (lda elems).
// lanes 0-15: M=lane, K 0-7 & 16-23 ; lanes 16-31: M=lane-16, K 8-15 & 24-31
DEVI v16bf load_a_bf(const __bf16* A, int row0, int k0, int lda, int lane) {
  int m  = row0 + (lane & 15);
  int kh = (lane >> 4) << 3;              // 0 or 8
  const __bf16* r = A + (size_t)m * lda + k0 + kh;
  Frag f;
  f.u[0] = *(const uint4*)(r);            // K kh..kh+7
  f.u[1] = *(const uint4*)(r + 16);       // K kh+16..kh+23
  return f.v;
}

// Same A fragment but source is f32 (probability matrix), convert on the fly.
DEVI v16bf load_a_f32(const float* A, int row0, int k0, int lda, int lane) {
  int m  = row0 + (lane & 15);
  int kh = (lane >> 4) << 3;
  const float* r = A + (size_t)m * lda + k0 + kh;
  v16bf a;
#pragma unroll
  for (int i = 0; i < 8; ++i) { a[i] = (__bf16)r[i]; a[8 + i] = (__bf16)r[16 + i]; }
  return a;
}

// B fragment, 32x16 bf16 with B[k,n] = Wsrc[n,k] (Wsrc row-major [N x K], ldw elems).
// lane n = lane&15 ; lanes 0-15 hold K 0-7 & 16-23, lanes 16-31 hold K 8-15 & 24-31
DEVI v16bf load_b_tr(const __bf16* Wsrc, int col0, int k0, int ldw, int lane) {
  int n  = col0 + (lane & 15);
  int kb = (lane >> 4) << 3;              // 0 or 8
  const __bf16* r = Wsrc + (size_t)n * ldw + k0 + kb;
  Frag f;
  f.u[0] = *(const uint4*)(r);
  f.u[1] = *(const uint4*)(r + 16);
  return f.v;
}

// B fragment from an LDS-staged [64 x 32] bf16 tile (row = n within block tile).
DEVI v16bf load_b_lds(const __bf16* sB, int col0, int lane) {
  int n  = col0 + (lane & 15);
  int kb = (lane >> 4) << 3;
  const __bf16* r = sB + n * 32 + kb;
  Frag f;
  f.u[0] = *(const uint4*)(r);
  f.u[1] = *(const uint4*)(r + 16);
  return f.v;
}

// ---------------------------------------------------------------------------
// Generic WMMA GEMM: out[m,n] = sum_k A[m,k] * W[n,k]
// B tile (64 cols x 32 k = 4KB) is double-buffered in LDS via async copies.
// ADT: 0 = A bf16, 1 = A f32 (converted)
// OUT: 0 = bf16 out, 1 = f32 out * scale, 2 = bf16 out + x (NCHW f32) fused
// BATCH: 0 = none; 1 = attention scores per (b,h); 2 = attention read per (b,h)
// Block: 256 threads = 8 waves -> 128(M) x 64(N) tile, wave = 32x32.
// ---------------------------------------------------------------------------
template <int ADT, int OUT, int BATCH>
__global__ __launch_bounds__(256) void gemm_wmma(
    const void* __restrict__ Ap, const __bf16* __restrict__ Bp, void* __restrict__ Op,
    int lda, int ldw, int ldo, int K, float scale, const float* __restrict__ xsrc) {
  __shared__ __bf16 smemB[2][64][32];

  const int tid  = threadIdx.x;
  const int lane = tid & 31;
  const int wave = tid >> 5;
  const int m0   = blockIdx.x * 128 + (wave & 3) * 32;
  const int nblk = blockIdx.y * 64;
  const int n0w  = (wave >> 2) * 32;      // wave's col offset inside block tile

  size_t aoff = 0, boff = 0, ooff = 0;
  if (BATCH == 1) {  // scores: A=Q[16384x768], B=K[16384x768], O=P[12,4096,4096]
    int z = blockIdx.z, b = z / 3, h = z % 3;
    aoff = (size_t)b * 4096 * lda + (size_t)h * 256;
    boff = (size_t)b * 4096 * ldw + (size_t)h * 256;
    ooff = (size_t)z * 4096 * ldo;
  } else if (BATCH == 2) {  // read: A=P f32, B=Vt[12,256,4096], O=read[16384x768]
    int z = blockIdx.z, b = z / 3, h = z % 3;
    aoff = (size_t)z * 4096 * lda;
    boff = (size_t)z * 256 * ldw;
    ooff = (size_t)b * 4096 * ldo + (size_t)h * 256;
  }

  const __bf16* A  = (const __bf16*)Ap + (ADT == 0 ? aoff : 0);
  const float*  Af = (const float*)Ap + (ADT == 1 ? aoff : 0);
  const __bf16* Bw = Bp + boff;
  __bf16* Ob = (__bf16*)Op;
  float*  Of = (float*)Op;

  // per-thread async-copy slot: row 0..63 of the B tile, 16B segment 0..3
  const int crow = tid >> 2;
  const int cseg = (tid & 3) * 8;  // bf16 offset within the 32-wide k slab

  auto issue = [&](int buf, int k0) {
    const __bf16* g = Bw + (size_t)(nblk + crow) * ldw + k0 + cseg;
    unsigned int l = (unsigned int)(uintptr_t)&smemB[buf][crow][cseg];
    async_b128(l, g);
  };

  v8f acc00 = zero8(), acc01 = zero8(), acc10 = zero8(), acc11 = zero8();

  issue(0, 0);  // prologue

  for (int k0 = 0; k0 < K; k0 += 32) {
    const int buf = (k0 >> 5) & 1;
    wait_async0();       // our async writes (incl. this buffer) are in LDS
    __syncthreads();     // ...and visible to / ordered with all waves
    if (k0 + 32 < K) issue(buf ^ 1, k0 + 32);  // overlap next slab with compute

    if (k0 + 32 < K) {   // prefetch next A slab
      if (ADT == 0) __builtin_prefetch(A + (size_t)(m0 + (lane & 15)) * lda + k0 + 32, 0, 1);
      else          __builtin_prefetch(Af + (size_t)(m0 + (lane & 15)) * lda + k0 + 32, 0, 1);
    }
    v16bf a0, a1;
    if (ADT == 0) {
      a0 = load_a_bf(A, m0,      k0, lda, lane);
      a1 = load_a_bf(A, m0 + 16, k0, lda, lane);
    } else {
      a0 = load_a_f32(Af, m0,      k0, lda, lane);
      a1 = load_a_f32(Af, m0 + 16, k0, lda, lane);
    }
    v16bf b0 = load_b_lds(&smemB[buf][0][0], n0w,      lane);
    v16bf b1 = load_b_lds(&smemB[buf][0][0], n0w + 16, lane);
    acc00 = wmma_bf16(a0, b0, acc00);
    acc01 = wmma_bf16(a0, b1, acc01);
    acc10 = wmma_bf16(a1, b0, acc10);
    acc11 = wmma_bf16(a1, b1, acc11);
  }

  // D layout: VGPR r -> M = r + (lane<16?0:8), N = lane&15
  const int nn = lane & 15;
  const int mo = (lane >> 4) << 3;
#pragma unroll
  for (int ti = 0; ti < 2; ++ti) {
#pragma unroll
    for (int tj = 0; tj < 2; ++tj) {
      v8f acc = (ti == 0) ? (tj == 0 ? acc00 : acc01) : (tj == 0 ? acc10 : acc11);
#pragma unroll
      for (int r = 0; r < 8; ++r) {
        int m = m0 + ti * 16 + mo + r;
        int n = nblk + n0w + tj * 16 + nn;
        float val = acc[r];
        size_t oi = ooff + (size_t)m * ldo + n;
        if (OUT == 1) {
          Of[oi] = val * scale;
        } else if (OUT == 0) {
          Ob[oi] = (__bf16)val;
        } else {  // OUT == 2 : + x (NCHW f32), m = b*4096 + s, n = channel
          int bb = m >> 12, s = m & 4095;
          float xv = xsrc[(((size_t)(bb * 256 + n)) << 12) | (size_t)s];
          Ob[oi] = (__bf16)(val + xv);
        }
      }
    }
  }
}

// ---------------------------------------------------------------------------
// 3x3 conv as 9-tap implicit GEMM over channel-last bf16 activations.
// in_cl [B,4096,256] bf16, wt [9,256(O),256(C)] bf16.
// OUTMODE 0: selu(acc+bias) -> bf16 channel-last ; 1: acc+bias+x -> f32 NCHW
// ---------------------------------------------------------------------------
DEVI v16bf load_a_conv(const __bf16* inb, int row0, int k0, int dy, int dx, int lane) {
  int m = row0 + (lane & 15);
  int kh = (lane >> 4) << 3;
  int y = (m >> 6) + dy, x = (m & 63) + dx;
  Frag f;
  if ((unsigned)y < 64u && (unsigned)x < 64u) {
    const __bf16* r = inb + ((size_t)(y * 64 + x)) * 256 + k0 + kh;
    f.u[0] = *(const uint4*)(r);
    f.u[1] = *(const uint4*)(r + 16);
  } else {
    f.u[0] = make_uint4(0, 0, 0, 0);
    f.u[1] = make_uint4(0, 0, 0, 0);
  }
  return f.v;
}

template <int OUTMODE>
__global__ __launch_bounds__(256) void conv3x3_wmma(
    const __bf16* __restrict__ in_cl, const __bf16* __restrict__ wt,
    const float* __restrict__ bias, const float* __restrict__ xsrc,
    float* __restrict__ out_f32, __bf16* __restrict__ out_bf) {
  const int lane = threadIdx.x & 31;
  const int wave = threadIdx.x >> 5;
  const int b  = blockIdx.z;
  const int m0 = blockIdx.x * 128 + (wave & 3) * 32;  // pixel in image
  const int n0 = blockIdx.y * 64  + (wave >> 2) * 32; // out channel

  const __bf16* inb = in_cl + (size_t)b * 4096 * 256;
  v8f acc00 = zero8(), acc01 = zero8(), acc10 = zero8(), acc11 = zero8();

  for (int t = 0; t < 9; ++t) {
    int dy = t / 3 - 1, dx = t % 3 - 1;
    const __bf16* wtt = wt + (size_t)t * 256 * 256;
#pragma unroll 2
    for (int k0 = 0; k0 < 256; k0 += 32) {
      v16bf a0 = load_a_conv(inb, m0,      k0, dy, dx, lane);
      v16bf a1 = load_a_conv(inb, m0 + 16, k0, dy, dx, lane);
      v16bf b0 = load_b_tr(wtt, n0,      k0, 256, lane);
      v16bf b1 = load_b_tr(wtt, n0 + 16, k0, 256, lane);
      acc00 = wmma_bf16(a0, b0, acc00);
      acc01 = wmma_bf16(a0, b1, acc01);
      acc10 = wmma_bf16(a1, b0, acc10);
      acc11 = wmma_bf16(a1, b1, acc11);
    }
  }

  const int nn = lane & 15;
  const int mo = (lane >> 4) << 3;
#pragma unroll
  for (int ti = 0; ti < 2; ++ti) {
#pragma unroll
    for (int tj = 0; tj < 2; ++tj) {
      v8f acc = (ti == 0) ? (tj == 0 ? acc00 : acc01) : (tj == 0 ? acc10 : acc11);
#pragma unroll
      for (int r = 0; r < 8; ++r) {
        int m = m0 + ti * 16 + mo + r;   // pixel
        int n = n0 + tj * 16 + nn;       // channel
        float u = acc[r] + bias[n];
        if (OUTMODE == 0) {
          float s = (u > 0.f) ? u : 1.6732632423543772f * (__expf(u) - 1.f);
          s *= 1.0507009873554805f;
          out_bf[(size_t)b * 4096 * 256 + (size_t)m * 256 + n] = (__bf16)s;
        } else {
          size_t oi = ((size_t)(b * 256 + n)) * 4096 + m;  // NCHW
          out_f32[oi] = u + xsrc[oi];
        }
      }
    }
  }
}

// ---------------------------------------------------------------------------
// Top-8 threshold + softmax, in place on one row of P [*, 4096].
// ---------------------------------------------------------------------------
__global__ __launch_bounds__(256) void topk_softmax_kernel(float* __restrict__ p) {
  const int L = 4096;
  float* row = p + (size_t)blockIdx.x * L;
  const int tid = threadIdx.x;

  float v[16];
  float t[8];
#pragma unroll
  for (int j = 0; j < 8; ++j) t[j] = -3.0e38f;
#pragma unroll
  for (int i = 0; i < 16; ++i) {
    float val = row[tid + i * 256];
    v[i] = val;
    if (val > t[7]) {
      t[7] = val;
#pragma unroll
      for (int j = 7; j > 0; --j)
        if (t[j] > t[j - 1]) { float tmp = t[j - 1]; t[j - 1] = t[j]; t[j] = tmp; }
    }
  }

  __shared__ float sm[256 * 8];
  __shared__ float red[256];
#pragma unroll
  for (int j = 0; j < 8; ++j) sm[tid * 8 + j] = t[j];
  __syncthreads();

  for (int s2 = 128; s2 >= 1; s2 >>= 1) {
    if (tid < s2) {
      float a[8], bl[8], o[8];
#pragma unroll
      for (int j = 0; j < 8; ++j) { a[j] = sm[tid * 8 + j]; bl[j] = sm[(tid + s2) * 8 + j]; }
      int ia = 0, ib = 0;
#pragma unroll
      for (int j = 0; j < 8; ++j) o[j] = (a[ia] >= bl[ib]) ? a[ia++] : bl[ib++];
#pragma unroll
      for (int j = 0; j < 8; ++j) sm[tid * 8 + j] = o[j];
    }
    __syncthreads();
  }
  const float thresh = sm[7];
  const float mx     = sm[0];

  float lsum = 0.f;
  float e[16];
#pragma unroll
  for (int i = 0; i < 16; ++i) {
    float ev = (v[i] < thresh) ? 0.f : __expf(v[i] - mx);
    e[i] = ev;
    lsum += ev;
  }
  red[tid] = lsum;
  __syncthreads();
  for (int s2 = 128; s2 >= 1; s2 >>= 1) {
    if (tid < s2) red[tid] += red[tid + s2];
    __syncthreads();
  }
  const float inv = 1.f / red[0];
#pragma unroll
  for (int i = 0; i < 16; ++i) row[tid + i * 256] = e[i] * inv;
}

// ---------------------------------------------------------------------------
// Packing / conversion helpers
// ---------------------------------------------------------------------------
__global__ void pack_cl_kernel(const float* __restrict__ in, __bf16* __restrict__ out, int total) {
  int i = blockIdx.x * 256 + threadIdx.x;
  if (i >= total) return;
  int c = i & 255, rest = i >> 8, s = rest & 4095, b = rest >> 12;
  out[i] = (__bf16)in[(((size_t)(b * 256 + c)) << 12) | (size_t)s];
}

__global__ void cvt_bf_kernel(const float* __restrict__ in, __bf16* __restrict__ out, int n) {
  int i = blockIdx.x * 256 + threadIdx.x;
  if (i < n) out[i] = (__bf16)in[i];
}

__global__ void pack_convw_kernel(const float* __restrict__ w, __bf16* __restrict__ wt, int total) {
  int i = blockIdx.x * 256 + threadIdx.x;
  if (i >= total) return;
  int c = i & 255, o = (i >> 8) & 255, t = i >> 16;  // wt[t][o][c] = w[o][c][t]
  wt[i] = (__bf16)w[(((size_t)(o * 256 + c)) * 9) + t];
}

__global__ void transpose_v_kernel(const __bf16* __restrict__ Vb, __bf16* __restrict__ Vt, int total) {
  int i = blockIdx.x * 256 + threadIdx.x;
  if (i >= total) return;
  int l = i & 4095, c = (i >> 12) & 255, bh = i >> 20, b = bh / 3, h = bh % 3;
  Vt[i] = Vb[((size_t)(b * 4096 + l)) * 768 + h * 256 + c];
}

// ---------------------------------------------------------------------------
extern "C" void kernel_launch(void* const* d_in, const int* in_sizes, int n_in,
                              void* d_out, int out_size, void* d_ws, size_t ws_size,
                              hipStream_t stream) {
  (void)in_sizes; (void)n_in; (void)out_size; (void)ws_size;
  const float* x   = (const float*)d_in[0];
  const float* ref = (const float*)d_in[1];
  const float* Wq  = (const float*)d_in[2];
  const float* Wk  = (const float*)d_in[3];
  const float* Wv  = (const float*)d_in[4];
  const float* Wo  = (const float*)d_in[5];
  const float* c1w = (const float*)d_in[6];
  const float* c1b = (const float*)d_in[7];
  const float* c2w = (const float*)d_in[8];
  const float* c2b = (const float*)d_in[9];
  float* out  = (float*)d_out;
  float* attn = out + (size_t)4 * 256 * 4096;  // [12,4096,4096] f32, in place

  // workspace carve-up (bf16 regions)
  char* ws = (char*)d_ws;
  size_t off = 0;
  auto carve = [&](size_t bytes) -> char* {
    char* p = ws + off;
    off += (bytes + 255) & ~(size_t)255;
    return p;
  };
  const size_t M = 16384;  // B*S
  __bf16* content = (__bf16*)carve(M * 256 * 2);
  __bf16* refs_cl = (__bf16*)carve(M * 256 * 2);
  __bf16* Wqb = (__bf16*)carve(768 * 256 * 2);
  __bf16* Wkb = (__bf16*)carve(768 * 256 * 2);
  __bf16* Wvb = (__bf16*)carve(768 * 256 * 2);
  __bf16* Wob = (__bf16*)carve(768 * 256 * 2);
  __bf16* wt1 = (__bf16*)carve((size_t)9 * 256 * 256 * 2);
  __bf16* wt2 = (__bf16*)carve((size_t)9 * 256 * 256 * 2);
  __bf16* Qb  = (__bf16*)carve(M * 768 * 2);
  __bf16* Kb  = (__bf16*)carve(M * 768 * 2);
  __bf16* Vb  = (__bf16*)carve(M * 768 * 2);
  __bf16* Vt  = (__bf16*)carve((size_t)12 * 256 * 4096 * 2);
  __bf16* readb = (__bf16*)carve(M * 768 * 2);
  __bf16* h_cl  = (__bf16*)carve(M * 256 * 2);
  __bf16* f1_cl = (__bf16*)carve(M * 256 * 2);

  const float scale = 1.f / sqrtf(768.f);

  // 1) pack activations + weights to bf16
  pack_cl_kernel<<<dim3(16384), dim3(256), 0, stream>>>(x,   content, 4194304);
  pack_cl_kernel<<<dim3(16384), dim3(256), 0, stream>>>(ref, refs_cl, 4194304);
  cvt_bf_kernel<<<dim3(768), dim3(256), 0, stream>>>(Wq, Wqb, 196608);
  cvt_bf_kernel<<<dim3(768), dim3(256), 0, stream>>>(Wk, Wkb, 196608);
  cvt_bf_kernel<<<dim3(768), dim3(256), 0, stream>>>(Wv, Wvb, 196608);
  cvt_bf_kernel<<<dim3(768), dim3(256), 0, stream>>>(Wo, Wob, 196608);
  pack_convw_kernel<<<dim3(2304), dim3(256), 0, stream>>>(c1w, wt1, 589824);
  pack_convw_kernel<<<dim3(2304), dim3(256), 0, stream>>>(c2w, wt2, 589824);

  // 2) projections: Q/K/V [16384 x 768]
  gemm_wmma<0, 0, 0><<<dim3(128, 12, 1), dim3(256), 0, stream>>>(
      content, Wqb, Qb, 256, 256, 768, 256, 1.f, nullptr);
  gemm_wmma<0, 0, 0><<<dim3(128, 12, 1), dim3(256), 0, stream>>>(
      refs_cl, Wkb, Kb, 256, 256, 768, 256, 1.f, nullptr);
  gemm_wmma<0, 0, 0><<<dim3(128, 12, 1), dim3(256), 0, stream>>>(
      refs_cl, Wvb, Vb, 256, 256, 768, 256, 1.f, nullptr);

  // 3) Vt [12, 256, 4096] for the read GEMM's contiguous B-fragments
  transpose_v_kernel<<<dim3(49152), dim3(256), 0, stream>>>(Vb, Vt, 12582912);

  // 4) scores P = Q K^T * scale, f32, written straight into attn_map region
  gemm_wmma<0, 1, 1><<<dim3(32, 64, 12), dim3(256), 0, stream>>>(
      Qb, Kb, attn, 768, 768, 4096, 256, scale, nullptr);

  // 5) top-8 mask + softmax in place (attn region becomes reference attn_map)
  topk_softmax_kernel<<<dim3(49152), dim3(256), 0, stream>>>(attn);

  // 6) read = P V  (A = f32 P, converted on the fly)
  gemm_wmma<1, 0, 2><<<dim3(32, 4, 12), dim3(256), 0, stream>>>(
      attn, Vt, readb, 4096, 4096, 768, 4096, 1.f, nullptr);

  // 7) output projection + residual: h = read Wo^T + x  (channel-last bf16)
  gemm_wmma<0, 2, 0><<<dim3(128, 4, 1), dim3(256), 0, stream>>>(
      readb, Wob, h_cl, 768, 768, 256, 768, 1.f, x);

  // 8) conv1 + SELU ; conv2 + bias + x -> final f32 NCHW output
  conv3x3_wmma<0><<<dim3(32, 4, 4), dim3(256), 0, stream>>>(
      h_cl, wt1, c1b, nullptr, nullptr, f1_cl);
  conv3x3_wmma<1><<<dim3(32, 4, 4), dim3(256), 0, stream>>>(
      f1_cl, wt2, c2b, x, out, nullptr);
}
